// STA2_Module_31920196944376
// MI455X (gfx1250) — matrix operand order repaired and checked by reading, not compiled, not run on hardware
//
#include <hip/hip_runtime.h>
#include <hip/hip_bf16.h>

typedef __attribute__((ext_vector_type(16))) __bf16 v16bf;
typedef __attribute__((ext_vector_type(8)))  float  v8f;

#define NB 4
#define NC 256
#define HW 4096
#define TQ 64     // queries per block
#define TJ 32     // keys per iteration

// LDS byte offsets (X region exactly overlaps Q+KV1+KV2 = 64KB union)
#define OFF_Q   0u        // 4 waves * 8 frags * 1KB   = 32KB  (A-frags of pre)
#define OFF_KV1 32768u    // 16 frags * 1KB            = 16KB  (B-frags, K=chan)
#define OFF_KV2 49152u    // 16 frags * 1KB            = 16KB  (B-frags, K=keys)
#define OFF_X   0u        // 4 waves * 16 frags * 1KB  = 64KB  (A-frags of [feat;cur])
#define OFF_P   65536u    // 4 waves * 1KB             = 4KB   (A-frags of P)
#define OFF_W   69632u    // 64 frags * 1KB            = 64KB  (B-frags of W chunk)
#define OFF_BN  135168u   // 256 f32 scale + 256 f32 add = 2KB
#define LDS_BYTES 137216u

__device__ __forceinline__ unsigned f2bfu(float x) {
  unsigned u = __float_as_uint(x);
  return (u + 0x7FFFu + ((u >> 16) & 1u)) >> 16;   // RNE f32->bf16
}
__device__ __forceinline__ unsigned pack2(float lo, float hi) {
  return f2bfu(lo) | (f2bfu(hi) << 16);
}
__device__ __forceinline__ float fexp(float x) {   // e^x via v_exp_f32
  return __builtin_amdgcn_exp2f(x * 1.4426950408889634f);
}
__device__ __forceinline__ float rowmax16(float v) {
  #pragma unroll
  for (int off = 8; off; off >>= 1) v = fmaxf(v, __shfl_xor(v, off, 16));
  return v;
}
__device__ __forceinline__ float rowsum16(float v) {
  #pragma unroll
  for (int off = 8; off; off >>= 1) v += __shfl_xor(v, off, 16);
  return v;
}

__global__ __launch_bounds__(128, 1) void sta2_flash_fused(
    const float* __restrict__ pre, const float* __restrict__ cur,
    const float* __restrict__ Wt,  const float* __restrict__ bconv,
    const float* __restrict__ gamma, const float* __restrict__ beta,
    const float* __restrict__ rmean, const float* __restrict__ rvar,
    const float* __restrict__ alpha, float* __restrict__ out)
{
  __shared__ __align__(128) unsigned char smem[LDS_BYTES];

  const int tid  = threadIdx.x;
  const int w    = tid >> 5;          // wave id 0..3
  const int lane = tid & 31;
  const int gl   = lane >> 4;         // lane group
  const int nn   = lane & 15;
  const int bIdx = blockIdx.y;
  const int i0   = blockIdx.x * TQ;   // block query base
  const int i0w  = i0 + w * 16;       // wave query base
  const size_t baseB = (size_t)bIdx * NC * HW;

  // ---- stage folded BN params: scale = g*rsqrt(var+eps); add = (b-mean)*scale+beta
  for (int o = tid; o < 256; o += 128) {
    float inv = gamma[o] * rsqrtf(rvar[o] + 1e-5f);
    *(float*)(smem + OFF_BN + 4u * o)          = inv;
    *(float*)(smem + OFF_BN + 1024u + 4u * o)  = (bconv[o] - rmean[o]) * inv + beta[o];
  }
  const float aslope = alpha[0];

  // ---- stage Q (pre at query positions) as bf16 A-fragments
  for (int qi = 0; qi < 32; ++qi) {
    int q  = tid + 128 * qi;          // 0..4095
    int c  = (q >> 5) * 2;            // even channel
    int i  = (q & 31) * 2;            // even query (local)
    const float* p0 = pre + baseB + (size_t)c * HW + i0 + i;
    float2 fa = *(const float2*)p0;          // [c][i], [c][i+1]
    float2 fb = *(const float2*)(p0 + HW);   // [c+1][i], [c+1][i+1]
    int wv = i >> 4, kc = c >> 5, cm = c & 31;
    int g = (cm >> 3) & 1, t = ((cm >> 4) << 3) + (cm & 7);
    int M = i & 15;
    unsigned a = OFF_Q + (unsigned)((wv * 8 + kc) * 1024) + (unsigned)(M + 16 * g) * 32 + t * 2;
    *(unsigned*)(smem + a)      = pack2(fa.x, fb.x);
    *(unsigned*)(smem + a + 32) = pack2(fa.y, fb.y);
  }
  __syncthreads();

  v16bf Aq[8];
  #pragma unroll
  for (int kc = 0; kc < 8; ++kc)
    Aq[kc] = *(const v16bf*)(smem + OFF_Q + (unsigned)((w * 8 + kc) * 1024) + lane * 32);

  v8f Oacc[16];
  { v8f z = {}; 
    #pragma unroll
    for (int t = 0; t < 16; ++t) Oacc[t] = z; }
  float mrow[8], lrow[8];
  #pragma unroll
  for (int r = 0; r < 8; ++r) { mrow[r] = -3.0e38f; lrow[r] = 0.f; }

  // ================= flash attention main loop over key tiles =================
  for (int j0 = 0; j0 < HW; j0 += TJ) {
    // cooperative stage of cur[:, j0:j0+32] in both B-fragment layouts
    for (int qi = 0; qi < 16; ++qi) {
      int q  = tid + 128 * qi;        // 0..2047
      int c  = (q >> 4) * 2;
      int jj = (q & 15) * 2;
      const float* p0 = cur + baseB + (size_t)c * HW + j0 + jj;
      float2 fa = *(const float2*)p0;
      float2 fb = *(const float2*)(p0 + HW);
      // layout 1: K=channels, N=keys (for S = Q^T K)
      int kc = c >> 5, cm = c & 31, g1 = (cm >> 3) & 1, t1 = ((cm >> 4) << 3) + (cm & 7);
      int nh = jj >> 4, n1 = jj & 15;
      unsigned a1 = OFF_KV1 + (unsigned)((kc * 2 + nh) * 1024) + (unsigned)(g1 * 16 + n1) * 32 + t1 * 2;
      *(unsigned*)(smem + a1)      = pack2(fa.x, fb.x);
      *(unsigned*)(smem + a1 + 32) = pack2(fa.y, fb.y);
      // layout 2: K=keys, N=channels (for O += P V^T)
      int nch = c >> 4, n2 = c & 15, g2 = (jj >> 3) & 1, t2 = ((jj >> 4) << 3) + (jj & 7);
      unsigned a2 = OFF_KV2 + (unsigned)(nch * 1024) + (unsigned)(g2 * 16 + n2) * 32 + t2 * 2;
      *(unsigned*)(smem + a2)      = pack2(fa.x, fa.y);
      *(unsigned*)(smem + a2 + 32) = pack2(fb.x, fb.y);
    }
    __syncthreads();

    // S tiles: 16 queries x 32 keys, K-depth 256
    v8f s0 = {}, s1 = {};
    #pragma unroll
    for (int kc = 0; kc < 8; ++kc) {
      v16bf kb0 = *(const v16bf*)(smem + OFF_KV1 + (unsigned)((kc * 2 + 0) * 1024) + lane * 32);
      v16bf kb1 = *(const v16bf*)(smem + OFF_KV1 + (unsigned)((kc * 2 + 1) * 1024) + lane * 32);
      s0 = __builtin_amdgcn_wmma_f32_16x16x32_bf16(false, Aq[kc], false, kb0, (short)0, s0, false, false);
      s1 = __builtin_amdgcn_wmma_f32_16x16x32_bf16(false, Aq[kc], false, kb1, (short)0, s1, false, false);
    }

    // online softmax per row (rows striped: M = r + 8*gl, col = nn)
    #pragma unroll
    for (int r = 0; r < 8; ++r) {
      float sa = s0[r] * 0.0625f;               // * c^-0.5
      float sb = s1[r] * 0.0625f;
      float mnew = fmaxf(mrow[r], rowmax16(fmaxf(sa, sb)));
      float resc = fexp(mrow[r] - mnew);
      float pa = fexp(sa - mnew);
      float pb = fexp(sb - mnew);
      lrow[r] = lrow[r] * resc + rowsum16(pa + pb);
      mrow[r] = mnew;
      #pragma unroll
      for (int nch = 0; nch < 16; ++nch) Oacc[nch][r] *= resc;
      // scatter P into wave-local A-fragment layout
      int M = r + 8 * gl;
      int j = nn, gp = (j >> 3) & 1, tp = ((j >> 4) << 3) + (j & 7);
      *(unsigned short*)(smem + OFF_P + (unsigned)(w * 1024) + (unsigned)(M + 16 * gp) * 32 + tp * 2)
          = (unsigned short)f2bfu(pa);
      j = 16 + nn; gp = (j >> 3) & 1; tp = ((j >> 4) << 3) + (j & 7);
      *(unsigned short*)(smem + OFF_P + (unsigned)(w * 1024) + (unsigned)(M + 16 * gp) * 32 + tp * 2)
          = (unsigned short)f2bfu(pb);
    }
    asm volatile("s_wait_dscnt 0" ::: "memory");   // wave-local LDS RAW fence

    v16bf Pf = *(const v16bf*)(smem + OFF_P + (unsigned)(w * 1024) + lane * 32);
    #pragma unroll
    for (int nch = 0; nch < 16; ++nch) {
      v16bf vb = *(const v16bf*)(smem + OFF_KV2 + (unsigned)(nch * 1024) + lane * 32);
      Oacc[nch] = __builtin_amdgcn_wmma_f32_16x16x32_bf16(false, Pf, false, vb, (short)0, Oacc[nch], false, false);
    }
    __syncthreads();
  }

  // ---- normalize: feat = O / l
  #pragma unroll
  for (int r = 0; r < 8; ++r) {
    float linv = 1.0f / lrow[r];
    #pragma unroll
    for (int nch = 0; nch < 16; ++nch) Oacc[nch][r] *= linv;
  }

  // ---- build X = [feat ; cur] A-fragments (reuses the 64KB main-loop region)
  #pragma unroll
  for (int nch = 0; nch < 16; ++nch) {
    int cch = nch * 16 + nn;
    int kc2 = cch >> 5, cm = cch & 31;
    int gx = (cm >> 3) & 1, tx = ((cm >> 4) << 3) + (cm & 7);
    unsigned basea = OFF_X + (unsigned)((w * 16 + kc2) * 1024) + tx * 2;
    #pragma unroll
    for (int r = 0; r < 8; ++r) {
      int M = r + 8 * gl;
      *(unsigned short*)(smem + basea + (unsigned)(M + 16 * gx) * 32) = (unsigned short)f2bfu(Oacc[nch][r]);
    }
  }
  for (int qi = 0; qi < 32; ++qi) {          // cur at query positions -> frags 8..15
    int q = tid + 128 * qi;
    int c = (q >> 5) * 2;
    int i = (q & 31) * 2;
    const float* p0 = cur + baseB + (size_t)c * HW + i0 + i;
    float2 fa = *(const float2*)p0;
    float2 fb = *(const float2*)(p0 + HW);
    int wv = i >> 4, kc = c >> 5, cm = c & 31;
    int g = (cm >> 3) & 1, t = ((cm >> 4) << 3) + (cm & 7);
    int M = i & 15;
    unsigned a = OFF_X + (unsigned)((wv * 16 + 8 + kc) * 1024) + (unsigned)(M + 16 * g) * 32 + t * 2;
    *(unsigned*)(smem + a)      = pack2(fa.x, fb.x);
    *(unsigned*)(smem + a + 32) = pack2(fa.y, fb.y);
  }
  __syncthreads();

  // ================= 1x1 conv + BN + PReLU over 64-wide output-channel chunks =================
  for (int o0 = 0; o0 < 256; o0 += 64) {
    for (int pi = 0; pi < 128; ++pi) {       // stage W[o0:o0+64, 0:512] as bf16 B-frags
      int p    = tid + 128 * pi;             // 0..16383
      int oo   = p >> 8;                     // 0..63
      int c512 = (p & 255) * 2;
      float2 fw = *(const float2*)(Wt + (size_t)(o0 + oo) * 512 + c512);
      int kc2 = c512 >> 5;
      int gw = (c512 >> 3) & 1, tw = (((c512 >> 4) & 1) << 3) + (c512 & 7);
      int nch2 = oo >> 4, n = oo & 15;
      unsigned a = OFF_W + (unsigned)((kc2 * 4 + nch2) * 1024) + (unsigned)(gw * 16 + n) * 32 + tw * 2;
      *(unsigned*)(smem + a) = pack2(fw.x, fw.y);
    }
    __syncthreads();

    #pragma unroll
    for (int nch2 = 0; nch2 < 4; ++nch2) {
      v8f acc = {};
      #pragma unroll
      for (int kc2 = 0; kc2 < 16; ++kc2) {
        v16bf xa = *(const v16bf*)(smem + OFF_X + (unsigned)((w * 16 + kc2) * 1024) + lane * 32);
        v16bf wb = *(const v16bf*)(smem + OFF_W + (unsigned)((kc2 * 4 + nch2) * 1024) + lane * 32);
        acc = __builtin_amdgcn_wmma_f32_16x16x32_bf16(false, xa, false, wb, (short)0, acc, false, false);
      }
      int o = o0 + nch2 * 16 + nn;
      float sc = *(const float*)(smem + OFF_BN + 4u * o);
      float ad = *(const float*)(smem + OFF_BN + 1024u + 4u * o);
      float v[8];
      #pragma unroll
      for (int r = 0; r < 8; ++r) {
        float y = acc[r] * sc + ad;
        v[r] = (y >= 0.f) ? y : aslope * y;
      }
      float* dst = out + ((size_t)(bIdx * 256 + o)) * HW + i0w + 8 * gl;
      *(float4*)dst       = make_float4(v[0], v[1], v[2], v[3]);
      *(float4*)(dst + 4) = make_float4(v[4], v[5], v[6], v[7]);
    }
    __syncthreads();
  }
}

extern "C" void kernel_launch(void* const* d_in, const int* in_sizes, int n_in,
                              void* d_out, int out_size, void* d_ws, size_t ws_size,
                              hipStream_t stream) {
  (void)in_sizes; (void)n_in; (void)d_ws; (void)ws_size; (void)out_size;
  const float* pre   = (const float*)d_in[0];
  const float* cur   = (const float*)d_in[1];
  const float* Wt    = (const float*)d_in[2];
  const float* bconv = (const float*)d_in[3];
  const float* gamma = (const float*)d_in[4];
  const float* beta  = (const float*)d_in[5];
  const float* rmean = (const float*)d_in[6];
  const float* rvar  = (const float*)d_in[7];
  const float* alpha = (const float*)d_in[8];
  float* out = (float*)d_out;
  dim3 grid(HW / TQ, NB, 1);   // 64 query tiles x 4 batches
  dim3 block(128, 1, 1);       // 4 wave32s
  sta2_flash_fused<<<grid, block, 0, stream>>>(pre, cur, Wt, bconv, gamma, beta,
                                               rmean, rvar, alpha, out);
}
